// SlotSetEncoder_52252572123213
// MI455X (gfx1250) — matrix operand order repaired and checked by reading, not compiled, not run on hardware
//
#include <hip/hip_runtime.h>
#include <hip/hip_bf16.h>
#include <math.h>
#include <stdint.h>

// ---------- types ----------
typedef __attribute__((ext_vector_type(16))) __bf16        v16bf;
typedef __attribute__((ext_vector_type(8)))  float         v8f;
typedef __attribute__((ext_vector_type(4))) unsigned int  v4u;
typedef __attribute__((ext_vector_type(8)))  unsigned int  v8u;
typedef __attribute__((ext_vector_type(4)))  float         v4f;
typedef __attribute__((ext_vector_type(4)))  unsigned int  u32x4;
typedef __attribute__((ext_vector_type(8)))  int           i32x8;
typedef __attribute__((ext_vector_type(4)))  int           i32x4;

#define N_BATCH   8
#define N_TOK     32768
#define D_IN      128
#define D_HAT     128
#define K_SLOT    64
#define H_DIM     128
#define EPSV      1e-8f
#define SCALEV    (0.08838834764831845f)   // 1/sqrt(128)

#define NTHREADS     512                    // 16 waves
#define WG_PER_BATCH 64
#define CHUNK_TOK    (N_TOK / WG_PER_BATCH) // 512
#define TILE_TOK     128
#define TILES_PER_WG (CHUNK_TOK / TILE_TOK) // 4

// ---------------- LDS layout (byte offsets) ----------------
// sWk   [128 cols][128 K] bf16            32768
// sWv   [128 cols][128 K] bf16            32768
// sQ    [64 slots][128 K] bf16            16384
// stag0 [128 tok][128 d]  f32 (TDM dest)  65536   (aliased by sM when current)
// stag1 [128 tok][128 d]  f32 (TDM dest)  65536
// sK    [128 tok][128 dh] bf16            32768   (A layout for M)
// sVT   [128 dh][128 tok] bf16            32768   (B layout for Shat)
// sWT   [64 slot][128 tok] bf16           16384   (A layout for Shat)
// sRinv 128 f32                             512
#define OFF_WK    0u
#define OFF_WV    32768u
#define OFF_Q     65536u
#define OFF_STAG0 81920u
#define OFF_STAG1 147456u
#define OFF_K     212992u
#define OFF_VT    245760u
#define OFF_WT    278528u
#define OFF_RINV  294912u
#define SMEM_BYTES 295424u

// ---------- WMMA fragment helpers ----------
// A fragment (16x32 bf16): lane L -> row=rowBase+(L&15); K = kb..kb+7, kb+16..kb+23
// with kb = kBase + (L>=16 ? 8 : 0).  src row-major [rows][ldk] bf16.
__device__ __forceinline__ v16bf load_a_frag(const unsigned short* base, int rowBase,
                                             int kBase, int ldk, int lane) {
  int row = rowBase + (lane & 15);
  int kb  = kBase + ((lane >> 4) << 3);
  const unsigned short* p = base + row * ldk + kb;
  union { struct { v4u lo, hi; } q; v16bf v; } cv;
  cv.q.lo = *(const v4u*)p;
  cv.q.hi = *(const v4u*)(p + 16);
  return cv.v;
}

// Same A fragment but sourced from f32 LDS with inline f32->bf16 convert.
__device__ __forceinline__ v16bf load_a_frag_f32(const float* base, int rowBase,
                                                 int kBase, int ldk, int lane) {
  int row = rowBase + (lane & 15);
  int kb  = kBase + ((lane >> 4) << 3);
  const float* p = base + row * ldk + kb;
  v4f f0 = *(const v4f*)p;
  v4f f1 = *(const v4f*)(p + 4);
  v4f f2 = *(const v4f*)(p + 16);
  v4f f3 = *(const v4f*)(p + 20);
  v16bf r;
  r[0]  = (__bf16)f0.x; r[1]  = (__bf16)f0.y; r[2]  = (__bf16)f0.z; r[3]  = (__bf16)f0.w;
  r[4]  = (__bf16)f1.x; r[5]  = (__bf16)f1.y; r[6]  = (__bf16)f1.z; r[7]  = (__bf16)f1.w;
  r[8]  = (__bf16)f2.x; r[9]  = (__bf16)f2.y; r[10] = (__bf16)f2.z; r[11] = (__bf16)f2.w;
  r[12] = (__bf16)f3.x; r[13] = (__bf16)f3.y; r[14] = (__bf16)f3.z; r[15] = (__bf16)f3.w;
  return r;
}

// B fragment (32x16 bf16): lane L -> col=colBase+(L&15); K = kb..kb+7, kb+16..kb+23
// with kb = kBase + (L>=16 ? 8 : 0).  src [cols][ldk] bf16, K contiguous.
__device__ __forceinline__ v16bf load_b_frag(const unsigned short* base, int colBase,
                                             int kBase, int ldk, int lane) {
  int col = colBase + (lane & 15);
  int kb  = kBase + ((lane >> 4) << 3);
  const unsigned short* p = base + col * ldk + kb;
  union { struct { v4u lo, hi; } q; v16bf v; } cv;
  cv.q.lo = *(const v4u*)p;
  cv.q.hi = *(const v4u*)(p + 16);
  return cv.v;
}

__device__ __forceinline__ v8f wmma_bf16(v16bf a, v16bf b, v8f c) {
  return __builtin_amdgcn_wmma_f32_16x16x32_bf16(false, a, false, b, (short)0, c,
                                                 false, false);
}

// ---------- Tensor Data Mover: one 128x128 f32 tile -> LDS ----------
// D# per CDNA5 ISA ch.8: group0 = {flags, lds_addr, global_addr, type=2},
// group1 = {data_size=4B, tensor_dim0=128, tensor_dim1=128, tile 128x128,
//           tensor_dim0_stride=128}.  Groups 2/3 (and trailing group) zero (2D).
__device__ __forceinline__ void tdm_load_tile(const float* gptr, unsigned lds_off) {
  unsigned long long ga = (unsigned long long)(uintptr_t)gptr;
  u32x4 g0;
  g0[0] = 1u;                                              // count=1, user mode
  g0[1] = lds_off;                                         // LDS byte address
  g0[2] = (unsigned)ga;                                    // global_addr[31:0]
  g0[3] = ((unsigned)(ga >> 32) & 0x01FFFFFFu) | 0x80000000u; // ga[56:32] | type=2
  i32x8 g1;
  g1[0] = (2 << 16);                                       // data_size = 4 bytes
  g1[1] = (D_IN << 16);                                    // tensor_dim0[15:0] @ [31:16]
  g1[2] = (TILE_TOK << 16);                                // td0 hi=0 | tensor_dim1 lo
  g1[3] = (D_IN << 16);                                    // td1 hi=0 | tile_dim0
  g1[4] = TILE_TOK;                                        // tile_dim1 | tile_dim2=0
  g1[5] = D_IN;                                            // tensor_dim0_stride lo32
  g1[6] = 0;                                               // stride hi | td1_stride lo
  g1[7] = 0;
  i32x4 gz4 = {0, 0, 0, 0};
  i32x8 gz8 = {0, 0, 0, 0, 0, 0, 0, 0};
  __builtin_amdgcn_tensor_load_to_lds(g0, g1, gz4, gz4, gz8, 0);
}

// ---------- pre-kernels ----------
__global__ void sse_zero_out(float* p, int nElem) {
  int i = blockIdx.x * 256 + threadIdx.x;
  if (i < nElem) p[i] = 0.0f;
}

// q[b,k,:] = S[b,k,:] @ Wq  -> bf16 in workspace [8][64][128]
__global__ void sse_qproj(const float* __restrict__ S, const float* __restrict__ Wq,
                          unsigned short* __restrict__ qbf) {
  int id  = blockIdx.x * 256 + threadIdx.x;   // 65536 total
  int b   = id >> 13;
  int rem = id & 8191;
  int k   = rem >> 7;
  int e   = rem & 127;
  const float* sp = S + ((size_t)b * K_SLOT + k) * H_DIM;
  float acc = 0.0f;
  #pragma unroll 8
  for (int h = 0; h < H_DIM; ++h) acc += sp[h] * Wq[h * D_HAT + e];
  ((__bf16*)qbf)[id] = (__bf16)acc;
}

// ---------- fused main kernel ----------
__global__ void __launch_bounds__(NTHREADS)
sse_main(const float* __restrict__ X, const unsigned short* __restrict__ qbf,
         const float* __restrict__ Wk, const float* __restrict__ Wv,
         float* __restrict__ out) {
  extern __shared__ __align__(16) char smem[];
  unsigned short* sWk  = (unsigned short*)(smem + OFF_WK);
  unsigned short* sWv  = (unsigned short*)(smem + OFF_WV);
  unsigned short* sQ   = (unsigned short*)(smem + OFF_Q);
  unsigned short* sK   = (unsigned short*)(smem + OFF_K);
  unsigned short* sVT  = (unsigned short*)(smem + OFF_VT);
  unsigned short* sWT  = (unsigned short*)(smem + OFF_WT);
  float*          sRinv = (float*)(smem + OFF_RINV);

  const int tid   = threadIdx.x;
  const int lane  = tid & 31;
  const int wave  = tid >> 5;
  const int b     = blockIdx.x >> 6;
  const int chunk = blockIdx.x & 63;
  const int hi8   = (lane >> 4) << 3;

  // ---- stage weights (transposed, K-contiguous) + q ----
  for (int e = tid; e < D_IN * D_HAT; e += NTHREADS) {
    int row = e >> 7, col = e & 127;
    ((__bf16*)sWk)[col * 128 + row] = (__bf16)Wk[e];
    ((__bf16*)sWv)[col * 128 + row] = (__bf16)Wv[e];
  }
  for (int e = tid; e < K_SLOT * D_HAT; e += NTHREADS)
    sQ[e] = qbf[(size_t)b * K_SLOT * D_HAT + e];

  const float* xbase = X + ((size_t)b * N_TOK + (size_t)chunk * CHUNK_TOK) * D_IN;

  // TDM prologue: kick tile 0 into staging buffer 0 (wave0 only; EXEC-independent)
  if (wave == 0) tdm_load_tile(xbase, OFF_STAG0);
  __syncthreads();   // weights staged (TDM not yet awaited)

  // work partition (16 waves):
  const int r2  = wave & 7;          // phase2/3 token-row block (8 x 16 rows)
  const int cb2 = (wave >> 3) * 64;  // phase2 d_hat half
  const int c3  = (wave >> 3) * 2;   // phase3 slot-tile pair
  const int rs  = wave & 3;          // phase4 slot-row block
  const int cb4 = (wave >> 2) * 32;  // phase4 d_hat col pair

  v8f sacc0 = {}, sacc1 = {};        // persistent S_hat accumulators

  for (int t = 0; t < TILES_PER_WG; ++t) {
    // ---- pipeline: issue TDM for tile t+1, wait for tile t ----
    if (wave == 0) {
      if (t + 1 < TILES_PER_WG) {
        tdm_load_tile(xbase + (size_t)(t + 1) * TILE_TOK * D_IN,
                      ((t + 1) & 1) ? OFF_STAG1 : OFF_STAG0);
        __builtin_amdgcn_s_wait_tensorcnt(1);   // oldest (tile t) complete
      } else {
        __builtin_amdgcn_s_wait_tensorcnt(0);
      }
    }
    __syncthreads();
    const float* sXf = (const float*)(smem + ((t & 1) ? OFF_STAG1 : OFF_STAG0));
    float*       sM  = (float*)(smem + ((t & 1) ? OFF_STAG1 : OFF_STAG0)); // alias

    // ---- phase 2: k = X@Wk, v = X@Wv  (A converted f32->bf16 in-reg) ----
    {
      v16bf aX0 = load_a_frag_f32(sXf, r2 * 16,  0, 128, lane);
      v16bf aX1 = load_a_frag_f32(sXf, r2 * 16, 32, 128, lane);
      v16bf aX2 = load_a_frag_f32(sXf, r2 * 16, 64, 128, lane);
      v16bf aX3 = load_a_frag_f32(sXf, r2 * 16, 96, 128, lane);
      #pragma unroll
      for (int c = 0; c < 4; ++c) {
        int colBase = cb2 + c * 16;
        v8f ak = {}, av = {};
        #pragma unroll
        for (int kk = 0; kk < 4; ++kk) {
          v16bf a  = (kk == 0) ? aX0 : (kk == 1) ? aX1 : (kk == 2) ? aX2 : aX3;
          ak = wmma_bf16(a, load_b_frag(sWk, colBase, kk * 32, 128, lane), ak);
          av = wmma_bf16(a, load_b_frag(sWv, colBase, kk * 32, 128, lane), av);
        }
        int col = colBase + (lane & 15);
        #pragma unroll
        for (int i = 0; i < 8; ++i) {
          int row = r2 * 16 + i + hi8;
          ((__bf16*)sK )[row * 128 + col] = (__bf16)ak[i];
          ((__bf16*)sVT)[col * 128 + row] = (__bf16)av[i];   // transposed
        }
      }
    }
    __syncthreads();   // X staging consumed; sM alias now safe to write

    // ---- phase 3: M = k @ q^T, sigmoid ----
    {
      v16bf aK0 = load_a_frag(sK, r2 * 16,  0, 128, lane);
      v16bf aK1 = load_a_frag(sK, r2 * 16, 32, 128, lane);
      v16bf aK2 = load_a_frag(sK, r2 * 16, 64, 128, lane);
      v16bf aK3 = load_a_frag(sK, r2 * 16, 96, 128, lane);
      #pragma unroll
      for (int cc = 0; cc < 2; ++cc) {
        int colBase = (c3 + cc) * 16;
        v8f acc = {};
        acc = wmma_bf16(aK0, load_b_frag(sQ, colBase,  0, 128, lane), acc);
        acc = wmma_bf16(aK1, load_b_frag(sQ, colBase, 32, 128, lane), acc);
        acc = wmma_bf16(aK2, load_b_frag(sQ, colBase, 64, 128, lane), acc);
        acc = wmma_bf16(aK3, load_b_frag(sQ, colBase, 96, 128, lane), acc);
        int col = colBase + (lane & 15);
        #pragma unroll
        for (int i = 0; i < 8; ++i) {
          int row = r2 * 16 + i + hi8;
          float m = acc[i] * SCALEV;
          sM[row * 64 + col] = 1.0f / (1.0f + __expf(-m)) + EPSV;
        }
      }
    }
    __syncthreads();

    if (tid < TILE_TOK) {              // normalize over 64 slots
      float s = 0.0f;
      #pragma unroll 8
      for (int j = 0; j < 64; ++j) s += sM[tid * 64 + j];
      sRinv[tid] = 1.0f / s;
    }
    __syncthreads();

    for (int e = tid; e < K_SLOT * TILE_TOK; e += NTHREADS) {   // W^T bf16, A layout
      int tok = e >> 6, slot = e & 63;
      ((__bf16*)sWT)[slot * 128 + tok] = (__bf16)(sM[tok * 64 + slot] * sRinv[tok]);
    }
    __syncthreads();

    // ---- phase 4: S_hat += W^T @ v (contract over 128 tokens) ----
    {
      #pragma unroll
      for (int kk = 0; kk < 4; ++kk) {
        v16bf aW = load_a_frag(sWT, rs * 16, kk * 32, 128, lane);
        sacc0 = wmma_bf16(aW, load_b_frag(sVT, cb4,      kk * 32, 128, lane), sacc0);
        sacc1 = wmma_bf16(aW, load_b_frag(sVT, cb4 + 16, kk * 32, 128, lane), sacc1);
      }
    }
    __syncthreads();   // protect sK/sVT/sWT reuse next tile
  }

  // ---- flush partial S_hat with fp32 atomics ----
  float* outb = out + (size_t)b * K_SLOT * D_HAT;
  int col0 = cb4 + (lane & 15);
  #pragma unroll
  for (int i = 0; i < 8; ++i) {
    int slot = rs * 16 + i + hi8;
    unsafeAtomicAdd(&outb[slot * D_HAT + col0],      sacc0[i]);
    unsafeAtomicAdd(&outb[slot * D_HAT + col0 + 16], sacc1[i]);
  }
}

// ---------- launcher ----------
extern "C" void kernel_launch(void* const* d_in, const int* in_sizes, int n_in,
                              void* d_out, int out_size, void* d_ws, size_t ws_size,
                              hipStream_t stream) {
  const float* X  = (const float*)d_in[0];
  const float* S  = (const float*)d_in[1];
  const float* Wk = (const float*)d_in[2];
  const float* Wq = (const float*)d_in[3];
  const float* Wv = (const float*)d_in[4];
  float* out = (float*)d_out;
  unsigned short* qbf = (unsigned short*)d_ws;   // 8*64*128 bf16 = 128 KB

  (void)in_sizes; (void)n_in; (void)out_size; (void)ws_size;

  (void)hipFuncSetAttribute((const void*)sse_main,
                            hipFuncAttributeMaxDynamicSharedMemorySize, SMEM_BYTES);

  sse_zero_out<<<(N_BATCH * K_SLOT * D_HAT + 255) / 256, 256, 0, stream>>>(
      out, N_BATCH * K_SLOT * D_HAT);
  sse_qproj<<<(N_BATCH * K_SLOT * D_HAT) / 256, 256, 0, stream>>>(S, Wq, qbf);
  sse_main<<<N_BATCH * WG_PER_BATCH, NTHREADS, SMEM_BYTES, stream>>>(
      X, qbf, Wk, Wv, out);
}
